// GNNModel_43104291783409
// MI455X (gfx1250) — compile-verified, matrix operand order checked by reading, not run on hardware
//
#include <hip/hip_runtime.h>
#include <hip/hip_bf16.h>

// ---------------------------------------------------------------------------
// GNN forward fully fused for gfx1250 (MI455X): all GEMMs via
// v_wmma_f32_16x16x32_f16 (f16 A/B, f32 accumulate). One workgroup of
// 256 threads (8 wave32) processes 16 graphs; all activations live in LDS.
// Weights are packed once per launch into WMMA B-fragment layout in d_ws.
// ---------------------------------------------------------------------------

typedef __attribute__((ext_vector_type(16))) _Float16 v16h;
typedef __attribute__((ext_vector_type(8)))  _Float16 v8h;
typedef __attribute__((ext_vector_type(8)))  float    v8f;
typedef __attribute__((ext_vector_type(4)))  float    v4f;

#define BQ   65536   // batch
#define NN   5       // nodes per graph
#define EE   8       // edges per graph
#define DD   128     // node hidden
#define DEE  32      // edge hidden
#define LL   4       // layers
#define PD_  256
#define M1_  256
#define M2_  64
#define GPB  16      // graphs per block

// ---- workspace layout (in _Float16 elements) ----
#define O_MSG1 0u                    // 4*288*128 = 147456
#define O_MSG2 147456u               // 4*128*128 =  65536
#define O_UPD1 212992u               // 4*256*128 = 131072
#define O_UPD2 344064u               // 4*128*128 =  65536
#define O_POOL 409600u               // 640*256   = 163840
#define O_MLP1 573440u               // 256*256   =  65536
#define O_MLP2 638976u               // 256*64    =  16384
// total 655360 halves = 1,310,720 bytes

// ---- LDS layout (bytes) ----
#define S_XF   0u        // float    x[16][5][128]        40960
#define S_XH   40960u    // _Float16 x[16][5][128]        20480
#define S_EA   61440u    // _Float16 ea[16][8][32]         8192
#define S_BUF  69632u    // _Float16 buf[16384]           32768  (m1 / u1 / p / q)
#define S_AGG  102400u   // float    agg[16][5][128]      40960  (also rbuf 16x64)
#define S_EIX  143360u   // int      eidx[16]                64
#define SMEM_BYTES 143424u

__device__ __forceinline__ float lrelu(float v) { return v >= 0.0f ? v : 0.01f * v; }

__device__ __forceinline__ v8f splat8(float v) {
    v8f c;
#pragma unroll
    for (int r = 0; r < 8; ++r) c[r] = v;
    return c;
}

// WMMA D = A(16x32 f16) * B(32x16 f16) + C(16x16 f32)
__device__ __forceinline__ v8f wmma16(v16h a, v16h b, v8f c) {
    return __builtin_amdgcn_wmma_f32_16x16x32_f16(false, a, false, b, (short)0, c,
                                                  false, false);
}

// Load B fragment from packed weights: per (ktile,ntile), 32 lanes x 16 halves.
__device__ __forceinline__ v16h ldB(const _Float16* pk, int kt, int nt, int ntiles,
                                    int lane) {
    return *(const v16h*)(pk + (((size_t)(kt * ntiles + nt) * 32) + lane) * 16);
}

// Build A fragment (16x32, f16) from a contiguous f16 row in LDS.
// Layout: lane m = lane&15; koff = (lane>>4)*8;
// halves a[0..7]  = row[kb0 + koff      .. +7]   (K = kb0+koff   .. +7)
// halves a[8..15] = row[kb0 + 16 + koff .. +7]   (K = kb0+16+koff.. +7)
// Both chunks are 16-byte aligned -> two ds_load_b128.
__device__ __forceinline__ v16h ldA_h(const _Float16* row, int kb0, int koff) {
    v8h lo = *(const v8h*)(row + kb0 + koff);
    v8h hi = *(const v8h*)(row + kb0 + 16 + koff);
    v16h a;
#pragma unroll
    for (int i = 0; i < 8; ++i) {
        a[i]     = lo[i];
        a[8 + i] = hi[i];
    }
    return a;
}

// Same, but from an f32 row (4x ds_load_b128 + convert).
__device__ __forceinline__ v16h ldA_f(const float* row, int kb0, int koff) {
    v4f f0 = *(const v4f*)(row + kb0 + koff);
    v4f f1 = *(const v4f*)(row + kb0 + koff + 4);
    v4f f2 = *(const v4f*)(row + kb0 + 16 + koff);
    v4f f3 = *(const v4f*)(row + kb0 + 16 + koff + 4);
    v16h a;
#pragma unroll
    for (int i = 0; i < 4; ++i) {
        a[i]      = (_Float16)f0[i];
        a[4 + i]  = (_Float16)f1[i];
        a[8 + i]  = (_Float16)f2[i];
        a[12 + i] = (_Float16)f3[i];
    }
    return a;
}

// ---------------------------------------------------------------------------
// Pack an f32 weight (K x N, K multiple of 32, N multiple of 16) into WMMA
// B-fragment layout: dst[((kt*ntiles+nt)*32 + lane)*16 + h].
// B layout mirrors A with n = lane&15.
// ---------------------------------------------------------------------------
__global__ void gnn_pack_b(const float* __restrict__ W, _Float16* __restrict__ dst,
                           int K, int N) {
    int idx = blockIdx.x * 256 + threadIdx.x;
    if (idx >= K * N) return;
    int h    = idx & 15;
    int lane = (idx >> 4) & 31;
    int t    = idx >> 9;               // kt*ntiles + nt
    int ntiles = N >> 4;
    int kt = t / ntiles;
    int nt = t - kt * ntiles;
    int n  = nt * 16 + (lane & 15);
    int kb = (h & 7) + (h < 8 ? 0 : 16) + ((lane >> 4) << 3);
    int k  = kt * 32 + kb;
    dst[idx] = (_Float16)W[(size_t)k * N + n];
}

// ---------------------------------------------------------------------------
// Fused forward: encoders + 4 message-passing layers + pool + MLP head + mean.
// ---------------------------------------------------------------------------
__global__ __launch_bounds__(256) void gnn_main(
    const float* __restrict__ node_feat, const float* __restrict__ edge_feat,
    const long long* __restrict__ edge_index,
    const float* __restrict__ node_w, const float* __restrict__ node_b,
    const float* __restrict__ edge_w, const float* __restrict__ edge_b,
    const _Float16* __restrict__ pk_msg1, const float* __restrict__ msg_b1,
    const _Float16* __restrict__ pk_msg2, const float* __restrict__ msg_b2,
    const _Float16* __restrict__ pk_upd1, const float* __restrict__ upd_b1,
    const float* __restrict__ upd_g1, const float* __restrict__ upd_be1,
    const _Float16* __restrict__ pk_upd2, const float* __restrict__ upd_b2,
    const float* __restrict__ upd_g2, const float* __restrict__ upd_be2,
    const _Float16* __restrict__ pk_pool, const float* __restrict__ pool_b,
    const float* __restrict__ pool_g, const float* __restrict__ pool_be,
    const _Float16* __restrict__ pk_mlp1, const float* __restrict__ mlp1_b,
    const float* __restrict__ mlp1_g, const float* __restrict__ mlp1_be,
    const _Float16* __restrict__ pk_mlp2, const float* __restrict__ mlp2_b,
    const float* __restrict__ mlp2_g, const float* __restrict__ mlp2_be,
    float* __restrict__ out) {
    extern __shared__ char smem[];
    float*    xs  = (float*)(smem + S_XF);
    _Float16* xh  = (_Float16*)(smem + S_XH);
    _Float16* eah = (_Float16*)(smem + S_EA);
    _Float16* buf = (_Float16*)(smem + S_BUF);
    float*    agg = (float*)(smem + S_AGG);
    int*      eis = (int*)(smem + S_EIX);   // [0..7]=src, [8..15]=dst

    const int tid  = threadIdx.x;
    const int wave = tid >> 5;
    const int lane = tid & 31;
    const int ln   = lane & 15;
    const int hi8  = (lane >> 4) << 3;   // +8 row/K offset for upper half-wave
    const int koff = hi8;
    const long long base = (long long)blockIdx.x * GPB;

    if (tid < 16) eis[tid] = (int)edge_index[tid];

    // ---- node encoder: x = node_feat @ node_w + node_b (K=7, scalar) ----
    for (int i = tid; i < GPB * NN * DD; i += 256) {
        int g = i / (NN * DD), rem = i - g * (NN * DD);
        int n = rem / DD, d = rem & (DD - 1);
        const float* nf = node_feat + ((size_t)(base + g) * NN + n) * 7;
        float v = node_b[d];
#pragma unroll
        for (int k = 0; k < 7; ++k) v += nf[k] * node_w[k * DD + d];
        xs[i] = v;
        xh[i] = (_Float16)v;
    }
    // ---- edge encoder: ea = edge_feat @ edge_w + edge_b (K=2) ----
    for (int i = tid; i < GPB * EE * DEE; i += 256) {
        int g = i >> 8, rem = i & 255;
        int e = rem >> 5, d = rem & 31;
        const float* ef = edge_feat + ((size_t)(base + g) * EE + e) * 2;
        float v = edge_b[d] + ef[0] * edge_w[d] + ef[1] * edge_w[DEE + d];
        eah[i] = (_Float16)v;
    }
    __syncthreads();

    // =================== message passing layers ===================
    for (int l = 0; l < LL; ++l) {
        for (int i = tid; i < GPB * NN * DD; i += 256) agg[i] = 0.0f;
        __syncthreads();

        // ---- msg1: [x_dst|x_src|ea](128x288) @ W(288x128), mtile = wave ----
        {
            const _Float16* pk  = pk_msg1 + (size_t)l * 288 * 128;
            __builtin_prefetch(pk + (size_t)lane * 64, 0, 1);
            const float* bias   = msg_b1 + l * DD;
            v8f acc[8];
#pragma unroll
            for (int nt = 0; nt < 8; ++nt) acc[nt] = splat8(bias[nt * 16 + ln]);
            int row = wave * 16 + ln, g = row >> 3, e = row & 7;
            const _Float16* xi = xh + (g * NN + eis[8 + e]) * DD;  // h_i = x[dst]
            const _Float16* xj = xh + (g * NN + eis[e]) * DD;      // h_j = x[src]
            const _Float16* er = eah + (g * EE + e) * DEE;
#pragma unroll
            for (int kt = 0; kt < 9; ++kt) {
                v16h a;
                if (kt < 4)      a = ldA_h(xi, kt * 32, koff);
                else if (kt < 8) a = ldA_h(xj, kt * 32 - 128, koff);
                else             a = ldA_h(er, 0, koff);
#pragma unroll
                for (int nt = 0; nt < 8; ++nt)
                    acc[nt] = wmma16(a, ldB(pk, kt, nt, 8, lane), acc[nt]);
            }
#pragma unroll
            for (int nt = 0; nt < 8; ++nt)
#pragma unroll
                for (int r = 0; r < 8; ++r) {
                    int mr = wave * 16 + r + hi8;
                    buf[mr * DD + nt * 16 + ln] = (_Float16)lrelu(acc[nt][r]);
                }
        }
        __syncthreads();

        // ---- msg2 (128x128 @ 128x128) + scatter-add into agg[dst] ----
        {
            const _Float16* pk = pk_msg2 + (size_t)l * 128 * 128;
            __builtin_prefetch(pk + (size_t)lane * 64, 0, 1);
            const float* bias  = msg_b2 + l * DD;
            v8f acc[8];
#pragma unroll
            for (int nt = 0; nt < 8; ++nt) acc[nt] = splat8(bias[nt * 16 + ln]);
            const _Float16* arow = buf + (wave * 16 + ln) * DD;
#pragma unroll
            for (int kt = 0; kt < 4; ++kt) {
                v16h a = ldA_h(arow, kt * 32, koff);
#pragma unroll
                for (int nt = 0; nt < 8; ++nt)
                    acc[nt] = wmma16(a, ldB(pk, kt, nt, 8, lane), acc[nt]);
            }
#pragma unroll
            for (int nt = 0; nt < 8; ++nt)
#pragma unroll
                for (int r = 0; r < 8; ++r) {
                    int row = wave * 16 + r + hi8, g = row >> 3, e = row & 7;
                    atomicAdd(&agg[(g * NN + eis[8 + e]) * DD + nt * 16 + ln],
                              lrelu(acc[nt][r]));
                }
        }
        __syncthreads();

        // ---- upd1: [x|agg](80x256) @ W(256x128); ntile = wave, mt loop ----
        // B fragments for this wave's n-tile are hoisted and reused over mt.
        {
            const _Float16* pk = pk_upd1 + (size_t)l * 256 * 128;
            const float* bias  = upd_b1 + l * DD;
            const float* g1    = upd_g1 + l * NN;
            const float* be1   = upd_be1 + l * NN;
            const int nt = wave;
            v16h bw[8];
#pragma unroll
            for (int kt = 0; kt < 8; ++kt) bw[kt] = ldB(pk, kt, nt, 8, lane);
            const float bv = bias[nt * 16 + ln];
            for (int mt = 0; mt < 5; ++mt) {
                v8f acc = splat8(bv);
                int row = mt * 16 + ln, g = row / NN, ni = row - g * NN;
                const _Float16* xrow = xh + (g * NN + ni) * DD;
                const float*    grow = agg + (g * NN + ni) * DD;
#pragma unroll
                for (int kt = 0; kt < 8; ++kt) {
                    v16h a = (kt < 4) ? ldA_h(xrow, kt * 32, koff)
                                      : ldA_f(grow, (kt - 4) * 32, koff);
                    acc = wmma16(a, bw[kt], acc);
                }
#pragma unroll
                for (int r = 0; r < 8; ++r) {
                    int rw = mt * 16 + r + hi8, gg = rw / NN, nn = rw - gg * NN;
                    float v = acc[r] * g1[nn] + be1[nn];
                    buf[rw * DD + nt * 16 + ln] = (_Float16)lrelu(v);
                }
            }
        }
        __syncthreads();

        // ---- upd2: u1(80x128) @ W(128x128); scale/lrelu; residual x += u ----
        {
            const _Float16* pk = pk_upd2 + (size_t)l * 128 * 128;
            const float* bias  = upd_b2 + l * DD;
            const float* g2    = upd_g2 + l * NN;
            const float* be2   = upd_be2 + l * NN;
            const int nt = wave;
            v16h bw[4];
#pragma unroll
            for (int kt = 0; kt < 4; ++kt) bw[kt] = ldB(pk, kt, nt, 8, lane);
            const float bv = bias[nt * 16 + ln];
            for (int mt = 0; mt < 5; ++mt) {
                v8f acc = splat8(bv);
                const _Float16* arow = buf + (mt * 16 + ln) * DD;
#pragma unroll
                for (int kt = 0; kt < 4; ++kt)
                    acc = wmma16(ldA_h(arow, kt * 32, koff), bw[kt], acc);
#pragma unroll
                for (int r = 0; r < 8; ++r) {
                    int rw = mt * 16 + r + hi8, gg = rw / NN, nn = rw - gg * NN;
                    float v = lrelu(acc[r] * g2[nn] + be2[nn]);
                    int idx = rw * DD + nt * 16 + ln;
                    float nx = xs[idx] + v;
                    xs[idx] = nx;
                    xh[idx] = (_Float16)nx;
                }
            }
        }
        __syncthreads();
    }

    // =================== pool: x.reshape(16,640) @ W(640x256) ===================
    {
        __builtin_prefetch(pk_pool + (size_t)lane * 64, 0, 1);
        v8f acc[2];
#pragma unroll
        for (int j = 0; j < 2; ++j) acc[j] = splat8(pool_b[(wave + j * 8) * 16 + ln]);
        const _Float16* arow = xh + ln * (NN * DD);   // graph row = ln
#pragma unroll
        for (int kt = 0; kt < 20; ++kt) {
            v16h a = ldA_h(arow, kt * 32, koff);
#pragma unroll
            for (int j = 0; j < 2; ++j)
                acc[j] = wmma16(a, ldB(pk_pool, kt, wave + j * 8, 16, lane), acc[j]);
        }
#pragma unroll
        for (int j = 0; j < 2; ++j)
#pragma unroll
            for (int r = 0; r < 8; ++r) {
                int c = (wave + j * 8) * 16 + ln, g = r + hi8;
                float v = lrelu(acc[j][r] * pool_g[c] + pool_be[c]);
                buf[g * PD_ + c] = (_Float16)v;   // p: 16x256 at buf[0..4095]
            }
    }
    __syncthreads();

    // =================== mlp1: p(16x256) @ W(256x256) ===================
    {
        v8f acc[2];
#pragma unroll
        for (int j = 0; j < 2; ++j) acc[j] = splat8(mlp1_b[(wave + j * 8) * 16 + ln]);
        const _Float16* arow = buf + ln * PD_;
#pragma unroll
        for (int kt = 0; kt < 8; ++kt) {
            v16h a = ldA_h(arow, kt * 32, koff);
#pragma unroll
            for (int j = 0; j < 2; ++j)
                acc[j] = wmma16(a, ldB(pk_mlp1, kt, wave + j * 8, 16, lane), acc[j]);
        }
#pragma unroll
        for (int j = 0; j < 2; ++j)
#pragma unroll
            for (int r = 0; r < 8; ++r) {
                int c = (wave + j * 8) * 16 + ln, g = r + hi8;
                float v = lrelu(acc[j][r] * mlp1_g[c] + mlp1_be[c]);
                buf[4096 + g * M1_ + c] = (_Float16)v;   // q: 16x256
            }
    }
    __syncthreads();

    // =================== mlp2: q(16x256) @ W(256x64), waves 0..3 ===================
    float* rb = agg;   // reuse as 16x64 f32
    if (wave < 4) {
        const int nt = wave;
        v8f acc = splat8(mlp2_b[nt * 16 + ln]);
        const _Float16* arow = buf + 4096 + ln * M1_;
#pragma unroll
        for (int kt = 0; kt < 8; ++kt)
            acc = wmma16(ldA_h(arow, kt * 32, koff), ldB(pk_mlp2, kt, nt, 4, lane),
                         acc);
#pragma unroll
        for (int r = 0; r < 8; ++r) {
            int g = r + hi8, c = nt * 16 + ln;
            rb[g * M2_ + c] = lrelu(acc[r] * mlp2_g[c] + mlp2_be[c]);
        }
    }
    __syncthreads();

    // ---- mean over M2=64 ----
    if (tid < GPB) {
        float s = 0.0f;
        for (int j = 0; j < M2_; ++j) s += rb[tid * M2_ + j];
        out[base + tid] = s * (1.0f / 64.0f);
    }
}

// ---------------------------------------------------------------------------
extern "C" void kernel_launch(void* const* d_in, const int* in_sizes, int n_in,
                              void* d_out, int out_size, void* d_ws, size_t ws_size,
                              hipStream_t stream) {
    (void)in_sizes; (void)n_in; (void)out_size; (void)ws_size;

    const float*     node_feat  = (const float*)d_in[0];
    const float*     edge_feat  = (const float*)d_in[1];
    const long long* edge_index = (const long long*)d_in[2];
    const float* node_w = (const float*)d_in[3];
    const float* node_b = (const float*)d_in[4];
    const float* edge_w = (const float*)d_in[5];
    const float* edge_b = (const float*)d_in[6];
    const float* msg_w1 = (const float*)d_in[7];
    const float* msg_b1 = (const float*)d_in[8];
    const float* msg_w2 = (const float*)d_in[9];
    const float* msg_b2 = (const float*)d_in[10];
    const float* upd_w1 = (const float*)d_in[11];
    const float* upd_b1 = (const float*)d_in[12];
    const float* upd_g1 = (const float*)d_in[13];
    const float* upd_be1 = (const float*)d_in[14];
    const float* upd_w2 = (const float*)d_in[15];
    const float* upd_b2 = (const float*)d_in[16];
    const float* upd_g2 = (const float*)d_in[17];
    const float* upd_be2 = (const float*)d_in[18];
    const float* pool_w = (const float*)d_in[19];
    const float* pool_b = (const float*)d_in[20];
    const float* pool_g = (const float*)d_in[21];
    const float* pool_be = (const float*)d_in[22];
    const float* mlp1_w = (const float*)d_in[23];
    const float* mlp1_b = (const float*)d_in[24];
    const float* mlp1_g = (const float*)d_in[25];
    const float* mlp1_be = (const float*)d_in[26];
    const float* mlp2_w = (const float*)d_in[27];
    const float* mlp2_b = (const float*)d_in[28];
    const float* mlp2_g = (const float*)d_in[29];
    const float* mlp2_be = (const float*)d_in[30];

    _Float16* ws = (_Float16*)d_ws;
    _Float16* pk_msg1 = ws + O_MSG1;
    _Float16* pk_msg2 = ws + O_MSG2;
    _Float16* pk_upd1 = ws + O_UPD1;
    _Float16* pk_upd2 = ws + O_UPD2;
    _Float16* pk_pool = ws + O_POOL;
    _Float16* pk_mlp1 = ws + O_MLP1;
    _Float16* pk_mlp2 = ws + O_MLP2;

    // Pack weights (layered tensors packed as one tall K; per-layer offsets
    // coincide because each layer's K is a multiple of 32).
    auto grid = [](int total) { return dim3((total + 255) / 256); };
    gnn_pack_b<<<grid(4 * 288 * 128), 256, 0, stream>>>(msg_w1, pk_msg1, 4 * 288, 128);
    gnn_pack_b<<<grid(4 * 128 * 128), 256, 0, stream>>>(msg_w2, pk_msg2, 4 * 128, 128);
    gnn_pack_b<<<grid(4 * 256 * 128), 256, 0, stream>>>(upd_w1, pk_upd1, 4 * 256, 128);
    gnn_pack_b<<<grid(4 * 128 * 128), 256, 0, stream>>>(upd_w2, pk_upd2, 4 * 128, 128);
    gnn_pack_b<<<grid(640 * 256), 256, 0, stream>>>(pool_w, pk_pool, 640, 256);
    gnn_pack_b<<<grid(256 * 256), 256, 0, stream>>>(mlp1_w, pk_mlp1, 256, 256);
    gnn_pack_b<<<grid(256 * 64), 256, 0, stream>>>(mlp2_w, pk_mlp2, 256, 64);

    gnn_main<<<BQ / GPB, 256, SMEM_BYTES, stream>>>(
        node_feat, edge_feat, edge_index, node_w, node_b, edge_w, edge_b,
        pk_msg1, msg_b1, pk_msg2, msg_b2,
        pk_upd1, upd_b1, upd_g1, upd_be1,
        pk_upd2, upd_b2, upd_g2, upd_be2,
        pk_pool, pool_b, pool_g, pool_be,
        pk_mlp1, mlp1_b, mlp1_g, mlp1_be,
        pk_mlp2, mlp2_b, mlp2_g, mlp2_be,
        (float*)d_out);
}
